// OptimizedBaselineAttention_29764123361372
// MI455X (gfx1250) — matrix operand order repaired and checked by reading, not compiled, not run on hardware
//
#include <hip/hip_runtime.h>

// ---------------------------------------------------------------------------
// MI455X (gfx1250) multi-head attention: bf16 WMMA GEMMs + flash attention.
// TDM (tensor_load_to_lds) feeds all LDS tiles (GEMM A/B, flash K/V) with
// hardware row padding; double-buffered against WMMA compute.
// B=4, S=2048, E=2048, H=16, D=128.  wave32, v_wmma_f32_16x16x32_bf16.
// ---------------------------------------------------------------------------

typedef unsigned short u16;
typedef unsigned int   u32;
typedef unsigned long long u64;
typedef __attribute__((ext_vector_type(16))) __bf16 v16bf;
typedef __attribute__((ext_vector_type(8)))  float  v8f;

union FragU { v16bf v; uint4 q[2]; };

__device__ __forceinline__ u16 f2bf(float f) {
  u32 u = __float_as_uint(f);
  u += 0x7FFFu + ((u >> 16) & 1u);      // round-to-nearest-even
  return (u16)(u >> 16);
}
__device__ __forceinline__ u32 pack2bf(float a, float b) {
  return (u32)f2bf(a) | ((u32)f2bf(b) << 16);
}

// A fragment (16x32 bf16, M rows with K contiguous):
//   lane-group g: elems 0..7 <- K=g*8+0..7 ; elems 8..15 <- K=16+g*8+0..7
__device__ __forceinline__ v16bf load_a_frag(const u16* rowPtr, int g) {
  FragU f;
  f.q[0] = *(const uint4*)(rowPtr + g * 8);
  f.q[1] = *(const uint4*)(rowPtr + 16 + g * 8);
  return f.v;
}
// B fragment (32x16 KxN from N rows with K contiguous):
//   lane-group g: elems 0..7 <- K=g*8+0..7 ; elems 8..15 <- K=16+g*8+0..7
__device__ __forceinline__ v16bf load_b_frag(const u16* rowPtr, int g) {
  FragU f;
  f.q[0] = *(const uint4*)(rowPtr + g * 8);
  f.q[1] = *(const uint4*)(rowPtr + 16 + g * 8);
  return f.v;
}

#define WMMA_BF16(A, B, C) \
  __builtin_amdgcn_wmma_f32_16x16x32_bf16(false, (A), false, (B), (short)0, (C), false, false)

#define LDSROW  40    // 32 bf16 + 8 pad  (80B rows, pad_interval=3, pad_amount=3)
#define LDSROWK 136   // 128 bf16 + 8 pad (272B rows, pad_interval=5, pad_amount=3)

// ---------------------------------------------------------------------------
// Tensor Data Mover: 2-D bf16 tile -> padded LDS buffer (D# per ISA ch.8).
// pad_interval code: 0=2DW,...,3=16DW(64B),5=64DW(256B); pad_amount code: 3=4DW.
// ---------------------------------------------------------------------------
#if __has_builtin(__builtin_amdgcn_tensor_load_to_lds) && \
    __has_builtin(__builtin_amdgcn_s_wait_tensorcnt)
#define USE_TDM 1
typedef __attribute__((ext_vector_type(4))) unsigned int v4u_t;
typedef __attribute__((ext_vector_type(8))) int v8i_t;
typedef __attribute__((ext_vector_type(4))) int v4i_t;

__device__ __forceinline__ void tdm_load_2d_bf16(
    u16* ldsDst, const u16* gsrc,
    u32 tensorW, u32 tensorH, u32 tileW, u32 tileH, u32 padI, u32 padA) {
  u32 lds = (u32)(size_t)ldsDst;               // LDS byte offset (addr[31:0])
  u64 ga  = (u64)(size_t)gsrc;
  v4u_t g0;
  g0[0] = 1u;                                  // count=1, user-mode, no gather
  g0[1] = lds;                                 // lds_addr
  g0[2] = (u32)ga;                             // global_addr[31:0]
  g0[3] = (u32)((ga >> 32) & 0x01FFFFFFull) | (2u << 30);  // addr[56:32]|type=2
  v8i_t g1;
  g1[0] = (int)((1u << 16)                     // data_size = 2 bytes
              | (1u << 20)                     // pad_enable
              | (padI << 22)                   // pad_interval
              | (padA << 25));                 // pad_amount
  g1[1] = (int)((tensorW & 0xFFFFu) << 16);    // tensor_dim0[15:0]
  g1[2] = (int)((tensorW >> 16) | ((tensorH & 0xFFFFu) << 16));
  g1[3] = (int)((tensorH >> 16) | (tileW << 16));
  g1[4] = (int)(tileH & 0xFFFFu);              // tile_dim1 (tile_dim2=0)
  g1[5] = (int)tensorW;                        // tensor_dim0_stride[31:0]
  g1[6] = 0;
  g1[7] = 0;
  v4i_t gz = {0, 0, 0, 0};
#if __clang_major__ >= 23
  v8i_t gz8 = {0, 0, 0, 0, 0, 0, 0, 0};
  __builtin_amdgcn_tensor_load_to_lds(g0, g1, gz, gz, gz8, 0);
#else
  __builtin_amdgcn_tensor_load_to_lds(g0, g1, gz, gz, 0);
#endif
}
#else
#define USE_TDM 0
#endif

// ---------------------------------------------------------------------------
// Shared GEMM core: acc[2][4] += A-tile * B^T-tile, bf16 operands.
// Block tile 128x128, k-step 32, 8 waves in 4x2 grid (32x64 per wave).
// TDM path: double-buffered LDS tiles fed by wave 0.
// ---------------------------------------------------------------------------
__device__ __forceinline__ void gemm_core(
    const u16* __restrict__ Ag, const u16* __restrict__ Bg,
    int mBase, int nBase, int Kdim, int Mtot, int Ntot,
    u16* As0, u16* Bs0, u16* As1, u16* Bs1, v8f acc[2][4]) {
  (void)Mtot; (void)Ntot; (void)As1; (void)Bs1;
  const int tid  = threadIdx.x;
  const int lane = tid & 31, wave = tid >> 5;
  const int wm = wave & 3, wn = wave >> 2;
  const int rsel = lane & 15, g = lane >> 4;

#if USE_TDM
  if (wave == 0) {
    tdm_load_2d_bf16(As0, Ag + (size_t)mBase * Kdim, (u32)Kdim, (u32)Mtot, 32, 128, 3, 3);
    tdm_load_2d_bf16(Bs0, Bg + (size_t)nBase * Kdim, (u32)Kdim, (u32)Ntot, 32, 128, 3, 3);
    __builtin_amdgcn_s_wait_tensorcnt(0);
  }
  __syncthreads();
  for (int k0 = 0; k0 < Kdim; k0 += 32) {
    u16* Ac = (k0 & 32) ? As1 : As0;
    u16* Bc = (k0 & 32) ? Bs1 : Bs0;
    u16* An = (k0 & 32) ? As0 : As1;
    u16* Bn = (k0 & 32) ? Bs0 : Bs1;
    const bool more = (k0 + 32) < Kdim;
    if (wave == 0 && more) {                   // prefetch next tiles via TDM
      tdm_load_2d_bf16(An, Ag + (size_t)mBase * Kdim + (k0 + 32),
                       (u32)Kdim, (u32)Mtot, 32, 128, 3, 3);
      tdm_load_2d_bf16(Bn, Bg + (size_t)nBase * Kdim + (k0 + 32),
                       (u32)Kdim, (u32)Ntot, 32, 128, 3, 3);
    }
    v16bf a[2], b[4];
    #pragma unroll
    for (int mt = 0; mt < 2; ++mt)
      a[mt] = load_a_frag(&Ac[(wm * 32 + mt * 16 + rsel) * LDSROW], g);
    #pragma unroll
    for (int nt = 0; nt < 4; ++nt)
      b[nt] = load_b_frag(&Bc[(wn * 64 + nt * 16 + rsel) * LDSROW], g);
    #pragma unroll
    for (int mt = 0; mt < 2; ++mt)
      #pragma unroll
      for (int nt = 0; nt < 4; ++nt)
        acc[mt][nt] = WMMA_BF16(a[mt], b[nt], acc[mt][nt]);
    if (wave == 0 && more) __builtin_amdgcn_s_wait_tensorcnt(0);
    __syncthreads();
  }
#else
  for (int k0 = 0; k0 < Kdim; k0 += 32) {
    uint4 ta[2], tb[2];
    #pragma unroll
    for (int i = 0; i < 2; ++i) {
      int chunk = tid + i * 256;
      int row = chunk >> 2, c = chunk & 3;
      ta[i] = *(const uint4*)(Ag + (size_t)(mBase + row) * Kdim + k0 + c * 8);
      tb[i] = *(const uint4*)(Bg + (size_t)(nBase + row) * Kdim + k0 + c * 8);
    }
    #pragma unroll
    for (int i = 0; i < 2; ++i) {
      int chunk = tid + i * 256;
      int row = chunk >> 2, c = chunk & 3;
      *(uint4*)(&As0[row * LDSROW + c * 8]) = ta[i];
      *(uint4*)(&Bs0[row * LDSROW + c * 8]) = tb[i];
    }
    __syncthreads();
    v16bf a[2], b[4];
    #pragma unroll
    for (int mt = 0; mt < 2; ++mt)
      a[mt] = load_a_frag(&As0[(wm * 32 + mt * 16 + rsel) * LDSROW], g);
    #pragma unroll
    for (int nt = 0; nt < 4; ++nt)
      b[nt] = load_b_frag(&Bs0[(wn * 64 + nt * 16 + rsel) * LDSROW], g);
    #pragma unroll
    for (int mt = 0; mt < 2; ++mt)
      #pragma unroll
      for (int nt = 0; nt < 4; ++nt)
        acc[mt][nt] = WMMA_BF16(a[mt], b[nt], acc[mt][nt]);
    __syncthreads();
  }
#endif
}

// ---------------------------------------------------------------------------
// Kernel 0: f32 -> bf16 streaming conversion (8 elems / thread).
// ---------------------------------------------------------------------------
__global__ __launch_bounds__(256) void cvt_f32_bf16_kernel(
    const float* __restrict__ src, u16* __restrict__ dst, int n8) {
  int i = blockIdx.x * 256 + threadIdx.x;
  if (i >= n8) return;
  size_t off = (size_t)i * 8;
  float4 a = *(const float4*)(src + off);
  float4 b = *(const float4*)(src + off + 4);
  uint4 o;
  o.x = pack2bf(a.x, a.y); o.y = pack2bf(a.z, a.w);
  o.z = pack2bf(b.x, b.y); o.w = pack2bf(b.z, b.w);
  *(uint4*)(dst + off) = o;
}

// ---------------------------------------------------------------------------
// Kernel 1: qkv = x @ W_qkv^T  (M=8192, N=6144, K=2048).
// A 128-wide N block never straddles a 2048 boundary, so the Q/K/V routing is
// block-uniform: hoisted to a scalar branch (no exec-mask divergence).
// ---------------------------------------------------------------------------
__global__ __launch_bounds__(256) void qkv_gemm_kernel(
    const u16* __restrict__ Xbf, const u16* __restrict__ Wbf,
    u16* __restrict__ Qo, u16* __restrict__ Ko, u16* __restrict__ Vto) {
  __shared__ u16 As0[128 * LDSROW], Bs0[128 * LDSROW];
  __shared__ u16 As1[128 * LDSROW], Bs1[128 * LDSROW];
  const int bm = blockIdx.x & 63;            // M/128 = 64
  const int bn = blockIdx.x >> 6;            // N/128 = 48
  const int mBase = bm * 128, nBase = bn * 128;
  const int lane = threadIdx.x & 31, wave = threadIdx.x >> 5;
  const int wm = wave & 3, wn = wave >> 2;
  const int rsel = lane & 15, g = lane >> 4;

  v8f zero = {};
  v8f acc[2][4];
  #pragma unroll
  for (int mt = 0; mt < 2; ++mt)
    #pragma unroll
    for (int nt = 0; nt < 4; ++nt) acc[mt][nt] = zero;

  gemm_core(Xbf, Wbf, mBase, nBase, 2048, 8192, 6144, As0, Bs0, As1, Bs1, acc);

  const int which = nBase >> 11;             // 0=Q 1=K 2=V (block-uniform)
  const int fBase = (nBase & 2047) + wn * 64;
  const float qscale = 0.08838834764831845f; // 1/sqrt(128)

  auto store_rowmajor = [&](u16* dst, float scale) {
    #pragma unroll
    for (int mt = 0; mt < 2; ++mt)
      #pragma unroll
      for (int nt = 0; nt < 4; ++nt)
        #pragma unroll
        for (int r = 0; r < 8; ++r) {
          int m = mBase + wm * 32 + mt * 16 + g * 8 + r;
          int f = fBase + nt * 16 + rsel;
          int bI = m >> 11, s = m & 2047;
          int h = f >> 7, d = f & 127;
          size_t bh = (size_t)bI * 16 + h;
          dst[(bh * 2048 + s) * 128 + d] = f2bf(acc[mt][nt][r] * scale);
        }
  };
  if (which == 0)      store_rowmajor(Qo, qscale);
  else if (which == 1) store_rowmajor(Ko, 1.0f);
  else {
    #pragma unroll
    for (int mt = 0; mt < 2; ++mt)
      #pragma unroll
      for (int nt = 0; nt < 4; ++nt)
        #pragma unroll
        for (int r = 0; r < 8; ++r) {
          int m = mBase + wm * 32 + mt * 16 + g * 8 + r;
          int f = fBase + nt * 16 + rsel;
          int bI = m >> 11, s = m & 2047;
          int h = f >> 7, d = f & 127;
          size_t bh = (size_t)bI * 16 + h;
          Vto[(bh * 128 + d) * 2048 + s] = f2bf(acc[mt][nt][r]);   // V^T
        }
  }
}

// ---------------------------------------------------------------------------
// Kernel 2: flash attention.  8 waves x 16 query rows = 128 rows per block.
// TDM double-buffers the K tile (32x128, 272B rows) and V^T tile (128x32,
// 80B rows) into LDS; loop body is ds_load_b128 -> v_wmma + softmax VALU.
// Grid: (B*H, S/128).
// ---------------------------------------------------------------------------
__global__ __launch_bounds__(256) void flash_attn_kernel(
    const u16* __restrict__ Q, const u16* __restrict__ Kd,
    const u16* __restrict__ Vt, u16* __restrict__ AO) {
#if USE_TDM
  __shared__ u16 Ks[2][32 * LDSROWK];
  __shared__ u16 Vs[2][128 * LDSROW];
#endif
  __shared__ u16 Ps[8][16 * LDSROW];
  const int tid  = threadIdx.x;
  const int lane = tid & 31, wave = tid >> 5;
  const int bh = blockIdx.x;                // 0..63
  const int b = bh >> 4, h = bh & 15;
  const int qBase = blockIdx.y * 128 + wave * 16;
  const int rsel = lane & 15, g = lane >> 4;

  const u16* q  = Q  + (size_t)bh * 2048 * 128;
  const u16* k  = Kd + (size_t)bh * 2048 * 128;
  const u16* vt = Vt + (size_t)bh * 128 * 2048;

  v16bf qf[4];
  #pragma unroll
  for (int ks = 0; ks < 4; ++ks)
    qf[ks] = load_a_frag(q + (size_t)(qBase + rsel) * 128 + ks * 32, g);

  v8f zero = {};
  v8f o[8];
  #pragma unroll
  for (int dt = 0; dt < 8; ++dt) o[dt] = zero;
  float ms[8], ls[8];
  #pragma unroll
  for (int r = 0; r < 8; ++r) { ms[r] = -1e30f; ls[r] = 0.0f; }

#if USE_TDM
  if (wave == 0) {
    tdm_load_2d_bf16(&Ks[0][0], k,  128, 2048, 128, 32, 5, 3);
    tdm_load_2d_bf16(&Vs[0][0], vt, 2048, 128, 32, 128, 3, 3);
    __builtin_amdgcn_s_wait_tensorcnt(0);
  }
  __syncthreads();
#endif

  for (int j = 0; j < 2048; j += 32) {
#if USE_TDM
    const int cur = (j >> 5) & 1;
    const bool more = (j + 32) < 2048;
    if (wave == 0 && more) {                  // prefetch next K/V tiles via TDM
      tdm_load_2d_bf16(&Ks[cur ^ 1][0], k + (size_t)(j + 32) * 128,
                       128, 2048, 128, 32, 5, 3);
      tdm_load_2d_bf16(&Vs[cur ^ 1][0], vt + (j + 32),
                       2048, 128, 32, 128, 3, 3);
    }
#endif
    // scores S = Q * K^T for keys [j, j+32)
    v8f s0 = zero, s1 = zero;
    #pragma unroll
    for (int ks = 0; ks < 4; ++ks) {
#if USE_TDM
      v16bf kf0 = load_b_frag(&Ks[cur][rsel        * LDSROWK + ks * 32], g);
      v16bf kf1 = load_b_frag(&Ks[cur][(16 + rsel) * LDSROWK + ks * 32], g);
#else
      v16bf kf0 = load_b_frag(k + (size_t)(j + rsel) * 128 + ks * 32, g);
      v16bf kf1 = load_b_frag(k + (size_t)(j + 16 + rsel) * 128 + ks * 32, g);
#endif
      s0 = WMMA_BF16(qf[ks], kf0, s0);
      s1 = WMMA_BF16(qf[ks], kf1, s1);
    }

    // Online softmax (row reductions across the 16-lane half).
    #pragma unroll
    for (int r = 0; r < 8; ++r) {
      float mx = fmaxf(s0[r], s1[r]);
      #pragma unroll
      for (int msk = 1; msk < 16; msk <<= 1)
        mx = fmaxf(mx, __shfl_xor(mx, msk, 32));
      float mnew  = fmaxf(ms[r], mx);
      float alpha = __expf(ms[r] - mnew);
      ms[r] = mnew;
      float p0 = __expf(s0[r] - mnew);
      float p1 = __expf(s1[r] - mnew);
      s0[r] = p0; s1[r] = p1;
      float rs = p0 + p1;
      #pragma unroll
      for (int msk = 1; msk < 16; msk <<= 1)
        rs += __shfl_xor(rs, msk, 32);
      ls[r] = ls[r] * alpha + rs;
      #pragma unroll
      for (int dt = 0; dt < 8; ++dt) o[dt][r] *= alpha;
    }

    // P repack C-layout -> A-fragment via per-wave LDS (DS in-order per wave).
    u16* ps = &Ps[wave][0];
    #pragma unroll
    for (int r = 0; r < 8; ++r) {
      int row = g * 8 + r;
      ps[row * LDSROW + rsel]      = f2bf(s0[r]);
      ps[row * LDSROW + 16 + rsel] = f2bf(s1[r]);
    }
    v16bf pf = load_a_frag(ps + rsel * LDSROW, g);

    // O += P (16x32) * V^T block (32x128)
    #pragma unroll
    for (int dt = 0; dt < 8; ++dt) {
#if USE_TDM
      v16bf vf = load_b_frag(&Vs[cur][(dt * 16 + rsel) * LDSROW], g);
#else
      v16bf vf = load_b_frag(vt + (size_t)(dt * 16 + rsel) * 2048 + j, g);
#endif
      o[dt] = WMMA_BF16(pf, vf, o[dt]);
    }
#if USE_TDM
    if (wave == 0 && more) __builtin_amdgcn_s_wait_tensorcnt(0);
    __syncthreads();
#endif
  }

  #pragma unroll
  for (int r = 0; r < 8; ++r) {
    float inv = 1.0f / ls[r];
    int srow = qBase + g * 8 + r;
    size_t rowoff = ((size_t)b * 2048 + srow) * 2048 + (size_t)h * 128;
    #pragma unroll
    for (int dt = 0; dt < 8; ++dt)
      AO[rowoff + dt * 16 + rsel] = f2bf(o[dt][r] * inv);
  }
}

// ---------------------------------------------------------------------------
// Kernel 3: out = attnOut @ W_out^T + b_out  (M=8192, N=2048, K=2048), f32 out
// ---------------------------------------------------------------------------
__global__ __launch_bounds__(256) void out_gemm_kernel(
    const u16* __restrict__ A, const u16* __restrict__ Wbf,
    const float* __restrict__ Bias, float* __restrict__ Out) {
  __shared__ u16 As0[128 * LDSROW], Bs0[128 * LDSROW];
  __shared__ u16 As1[128 * LDSROW], Bs1[128 * LDSROW];
  const int bm = blockIdx.x & 63;            // M/128 = 64
  const int bn = blockIdx.x >> 6;            // N/128 = 16
  const int mBase = bm * 128, nBase = bn * 128;
  const int lane = threadIdx.x & 31, wave = threadIdx.x >> 5;
  const int wm = wave & 3, wn = wave >> 2;
  const int rsel = lane & 15, g = lane >> 4;

  v8f zero = {};
  v8f acc[2][4];
  #pragma unroll
  for (int mt = 0; mt < 2; ++mt)
    #pragma unroll
    for (int nt = 0; nt < 4; ++nt) acc[mt][nt] = zero;

  gemm_core(A, Wbf, mBase, nBase, 2048, 8192, 2048, As0, Bs0, As1, Bs1, acc);

  #pragma unroll
  for (int mt = 0; mt < 2; ++mt) {
    #pragma unroll
    for (int nt = 0; nt < 4; ++nt) {
      int n = nBase + wn * 64 + nt * 16 + rsel;
      float bv = Bias[n];
      #pragma unroll
      for (int r = 0; r < 8; ++r) {
        int m = mBase + wm * 32 + mt * 16 + g * 8 + r;
        Out[(size_t)m * 2048 + n] = acc[mt][nt][r] + bv;
      }
    }
  }
}

// ---------------------------------------------------------------------------
// Host launcher.  Workspace (bf16 u16 elements):
//   Xbf[8192,2048] | Wqkvbf[6144,2048] | Woutbf[2048,2048] |
//   Q,K [B*H,S,D]  | V^T [B*H,D,S]     | AO [B,S,E]
// ---------------------------------------------------------------------------
extern "C" void kernel_launch(void* const* d_in, const int* in_sizes, int n_in,
                              void* d_out, int out_size, void* d_ws, size_t ws_size,
                              hipStream_t stream) {
  (void)in_sizes; (void)n_in; (void)out_size; (void)ws_size;
  const float* x    = (const float*)d_in[0];
  const float* Wqkv = (const float*)d_in[1];
  const float* Wout = (const float*)d_in[2];
  const float* bout = (const float*)d_in[3];
  float* out = (float*)d_out;

  const size_t N_X    = (size_t)8192 * 2048;
  const size_t N_WQKV = (size_t)6144 * 2048;
  const size_t N_WOUT = (size_t)2048 * 2048;
  const size_t N_QKV  = (size_t)64 * 2048 * 128;

  u16* Xbf = (u16*)d_ws;
  u16* Wqb = Xbf + N_X;
  u16* Wob = Wqb + N_WQKV;
  u16* Q   = Wob + N_WOUT;
  u16* Kb  = Q   + N_QKV;
  u16* Vt  = Kb  + N_QKV;
  u16* AO  = Vt  + N_QKV;

  cvt_f32_bf16_kernel<<<dim3((u32)(N_X    / 2048)), dim3(256), 0, stream>>>(x,    Xbf, (int)(N_X    / 8));
  cvt_f32_bf16_kernel<<<dim3((u32)(N_WQKV / 2048)), dim3(256), 0, stream>>>(Wqkv, Wqb, (int)(N_WQKV / 8));
  cvt_f32_bf16_kernel<<<dim3((u32)(N_WOUT / 2048)), dim3(256), 0, stream>>>(Wout, Wob, (int)(N_WOUT / 8));

  qkv_gemm_kernel<<<dim3(64 * 48), dim3(256), 0, stream>>>(Xbf, Wqb, Q, Kb, Vt);
  flash_attn_kernel<<<dim3(64, 16), dim3(256), 0, stream>>>(Q, Kb, Vt, AO);
  out_gemm_kernel<<<dim3(64 * 16), dim3(256), 0, stream>>>(AO, Wob, bout, out);
}
